// AttentionWithNoDeconvPos_72378788872224
// MI455X (gfx1250) — matrix operand order repaired and checked by reading, not compiled, not run on hardware
//
#include <hip/hip_runtime.h>
#include <hip/hip_bf16.h>

// ---------------------------------------------------------------------------
// Types for CDNA5 WMMA fragments (wave32)
// ---------------------------------------------------------------------------
typedef __attribute__((ext_vector_type(16))) _Float16 v16h;
typedef __attribute__((ext_vector_type(8)))  _Float16 v8h;
typedef __attribute__((ext_vector_type(8)))  float    v8f;

static __device__ __forceinline__ v8f wmma_f16(v16h a, v16h b, v8f c) {
  // D = A(16x32 f16) * B(32x16 f16) + C(16x16 f32)
  return __builtin_amdgcn_wmma_f32_16x16x32_f16(
      /*neg_a=*/false, a, /*neg_b=*/false, b,
      /*c_mod=*/(short)0, c, /*reuse_a=*/false, /*reuse_b=*/false);
}

// A-fragment (16x32 f16, row-major source): lane holds row = lane&15,
// K = hf*16 .. hf*16+15 (contiguous, matching load_b_frag)
static __device__ __forceinline__ v16h load_a_frag(const _Float16* rowp, int hf) {
  return *(const v16h*)(rowp + hf * 16);
}

// B-fragment (32x16 f16): lane holds column n = lane&15, K = hf*16..hf*16+15,
// contiguous when the source is stored K-major per column (i.e. W transposed).
static __device__ __forceinline__ v16h load_b_frag(const _Float16* colp, int hf) {
  return *(const v16h*)(colp + hf * 16);
}

// ---------------------------------------------------------------------------
// Problem constants
// ---------------------------------------------------------------------------
#define BB 32
#define NN 512
#define DD 512
#define HH 8
#define DHH 64
#define MROWS (BB * NN)   // 16384

// ---------------------------------------------------------------------------
// Kernel 1: fp32 [in][out] weight -> fp16 [out][in]  (512x512)
// ---------------------------------------------------------------------------
__global__ __launch_bounds__(256) void transpose_w_kernel(
    const float* __restrict__ W, _Float16* __restrict__ Wt) {
  int idx = blockIdx.x * 256 + threadIdx.x;   // 262144 = 512*512
  int o = idx >> 9, i = idx & 511;
  Wt[(size_t)o * DD + i] = (_Float16)W[(size_t)i * DD + o];
}

// ---------------------------------------------------------------------------
// Kernel 2: embedding  h[b,n,d] = x[b,n]*W_emb[n,d] + b_emb[n,d]  -> fp16
// ---------------------------------------------------------------------------
__global__ __launch_bounds__(256) void embed_kernel(
    const float* __restrict__ x, const float* __restrict__ W_emb,
    const float* __restrict__ b_emb, _Float16* __restrict__ Hh) {
  size_t idx = (size_t)blockIdx.x * 256 + threadIdx.x;  // B*N*D = 8388608
  int d = (int)(idx & 511);
  int row = (int)(idx >> 9);       // b*N + n
  int n = row & 511;
  int bidx = row >> 9;
  float v = x[bidx * NN + n] * W_emb[(size_t)n * DD + d] + b_emb[(size_t)n * DD + d];
  Hh[idx] = (_Float16)v;
}

// ---------------------------------------------------------------------------
// Kernel 3: WMMA GEMM  C[16384x512] = A(f16) @ Wt^T + bias
//   block = 256 threads (8 waves); block handles 16 rows x 512 cols,
//   wave w handles cols [w*64, w*64+64).  K loop: 16 x (1 a-frag, 4 WMMA).
// mode 0: out=q fp16 [b,h,n,dh], value scaled (1/sqrt(DH) folded in)
// mode 1: out=k fp16 [b,h,n,dh]
// mode 2: out=v fp16 transposed [b,h,dh,n]
// mode 3: out fp32 row-major [row,col]
// ---------------------------------------------------------------------------
__global__ __launch_bounds__(256) void wmma_gemm_qkvo(
    const _Float16* __restrict__ A, const _Float16* __restrict__ Wt,
    const float* __restrict__ bias, void* __restrict__ out,
    int mode, float scale) {
  const int K = DD;
  int wave = threadIdx.x >> 5;   // 0..7
  int lane = threadIdx.x & 31;
  int lm = lane & 15, hf = lane >> 4;
  int row0 = blockIdx.x * 16;
  int col0 = wave * 64;

  v8f c[4] = {};
  const _Float16* ap = A + (size_t)(row0 + lm) * K;
  for (int k0 = 0; k0 < K; k0 += 32) {
    if (k0 + 64 < K) __builtin_prefetch(ap + k0 + 64, 0, 3);  // global_prefetch_b8
    v16h a = load_a_frag(ap + k0, hf);
#pragma unroll
    for (int j = 0; j < 4; ++j) {
      const _Float16* bp = Wt + (size_t)(col0 + j * 16 + lm) * K + k0;
      v16h b = load_b_frag(bp, hf);
      c[j] = wmma_f16(a, b, c[j]);
    }
  }
#pragma unroll
  for (int j = 0; j < 4; ++j) {
    int col = col0 + j * 16 + lm;
    float bv = bias[col];
#pragma unroll
    for (int r = 0; r < 8; ++r) {
      int row = row0 + r + hf * 8;
      float v = (c[j][r] + bv) * scale;
      if (mode == 3) {
        ((float*)out)[(size_t)row * DD + col] = v;
      } else {
        int bi = row >> 9, n = row & 511;
        int hi = col >> 6, dh = col & 63;
        if (mode == 2)  // V stored transposed: vt[(b*H+h)*DH + dh][n]
          ((_Float16*)out)[(((size_t)bi * HH + hi) * DHH + dh) * NN + n] = (_Float16)v;
        else            // Q/K: [(b*H+h)*N + n][dh]
          ((_Float16*)out)[(((size_t)bi * HH + hi) * NN + n) * DHH + dh] = (_Float16)v;
      }
    }
  }
}

// ---------------------------------------------------------------------------
// Kernel 4: attention. One wave per (b, h, 16-row tile of queries).
//   scores (16x512) via 64 WMMA -> LDS -> softmax -> P fp16 in LDS
//   -> O = P @ V via 64 WMMA -> o_mid fp16 [B*N, D]
//   q was pre-scaled by 1/sqrt(DH); V is stored transposed so both matmuls
//   read contiguous B-fragments.
// ---------------------------------------------------------------------------
__global__ __launch_bounds__(32) void attention_kernel(
    const _Float16* __restrict__ q, const _Float16* __restrict__ k,
    const _Float16* __restrict__ vt, _Float16* __restrict__ o_mid) {
  __shared__ __align__(32) float    S[16][NN + 4];
  __shared__ __align__(32) _Float16 P[16][NN + 16];

  int lane = threadIdx.x;
  int lm = lane & 15, hf = lane >> 4;
  int nt = blockIdx.x & 31;       // query row-tile
  int bh = blockIdx.x >> 5;       // b*H + h
  const _Float16* qb = q + (size_t)bh * NN * DHH;
  const _Float16* kb = k + (size_t)bh * NN * DHH;
  const _Float16* vb = vt + (size_t)bh * DHH * NN;

  // Q fragments for both 32-wide d-chunks (DH = 64)
  const _Float16* qrow = qb + (size_t)(nt * 16 + lm) * DHH;
  v16h a0 = load_a_frag(qrow, hf);
  v16h a1 = load_a_frag(qrow + 32, hf);

  // scores = Q K^T (already includes 1/sqrt(DH) via q scaling)
  for (int mt = 0; mt < 32; ++mt) {
    const _Float16* krow = kb + (size_t)(mt * 16 + lm) * DHH;
    v16h b0 = load_b_frag(krow, hf);
    v16h b1 = load_b_frag(krow + 32, hf);
    v8f acc = {};
    acc = wmma_f16(a0, b0, acc);
    acc = wmma_f16(a1, b1, acc);
#pragma unroll
    for (int r = 0; r < 8; ++r) S[r + hf * 8][mt * 16 + lm] = acc[r];
  }
  __syncthreads();

  // softmax over 512 columns; lane pair (l, l^16) shares row = lane&15
  {
    int row = lm;
    int base = hf * 256;
    float mx = -3.0e38f;
    for (int c0 = 0; c0 < 256; ++c0) mx = fmaxf(mx, S[row][base + c0]);
    mx = fmaxf(mx, __shfl_xor(mx, 16, 32));
    float sum = 0.f;
    for (int c0 = 0; c0 < 256; ++c0) {
      float e = __expf(S[row][base + c0] - mx);
      S[row][base + c0] = e;
      sum += e;
    }
    sum += __shfl_xor(sum, 16, 32);
    float inv = 1.f / sum;
    for (int c0 = 0; c0 < 256; ++c0)
      P[row][base + c0] = (_Float16)(S[row][base + c0] * inv);
  }
  __syncthreads();

  // O = P @ V  (contraction over 512 context positions)
  int bi = bh >> 3, hi = bh & 7;
  for (int dt = 0; dt < 4; ++dt) {
    v8f acc = {};
    for (int m0 = 0; m0 < NN; m0 += 32) {
      v16h a = load_a_frag(&P[lm][m0], hf);
      v16h b = load_b_frag(vb + (size_t)(dt * 16 + lm) * NN + m0, hf);
      acc = wmma_f16(a, b, acc);
    }
    int col = hi * DHH + dt * 16 + lm;
#pragma unroll
    for (int r = 0; r < 8; ++r) {
      int n = nt * 16 + r + hf * 8;
      o_mid[((size_t)bi * NN + n) * DD + col] = (_Float16)acc[r];
    }
  }
}

// ---------------------------------------------------------------------------
// Kernel 5: unembed  x_atten[b,n] = dot(att[b,n,:], W_un[n,:]) + b_un[n]
// ---------------------------------------------------------------------------
__global__ __launch_bounds__(256) void unembed_kernel(
    const float* __restrict__ att, const float* __restrict__ W_un,
    const float* __restrict__ b_un, float* __restrict__ x_atten) {
  int idx = blockIdx.x * 256 + threadIdx.x;   // 16384 = B*N
  int n = idx & 511;
  const float* ap = att + (size_t)idx * DD;
  const float* wp = W_un + (size_t)n * DD;
  float acc = 0.f;
  for (int d = 0; d < DD; ++d) acc += ap[d] * wp[d];
  x_atten[idx] = acc + b_un[n];
}

// ---------------------------------------------------------------------------
// Kernel 6: small fp32 GEMM for the ff chain (tiny FLOP count)
// ---------------------------------------------------------------------------
__global__ __launch_bounds__(256) void ff_gemm_kernel(
    const float* __restrict__ A, const float* __restrict__ W,
    const float* __restrict__ bias, float* __restrict__ C,
    int M, int K, int Ncols) {
  int idx = blockIdx.x * 256 + threadIdx.x;
  if (idx >= M * Ncols) return;
  int m = idx / Ncols, nc = idx - m * Ncols;
  float acc = 0.f;
  for (int kk = 0; kk < K; ++kk) acc += A[(size_t)m * K + kk] * W[(size_t)kk * Ncols + nc];
  C[idx] = acc + bias[nc];
}

// ---------------------------------------------------------------------------
// Kernel 7: bilinear 8x32 -> 16x64 (half-pixel centers) + min/max normalize
// ---------------------------------------------------------------------------
__global__ __launch_bounds__(256) void resize_norm_kernel(
    const float* __restrict__ y3, float* __restrict__ yout) {
  __shared__ float buf[1024];
  __shared__ float rmn[256], rmx[256];
  int bidx = blockIdx.x;
  const float* src = y3 + bidx * 256;   // viewed as [8][32]

  for (int t = threadIdx.x; t < 1024; t += 256) {
    int i = t >> 6, j = t & 63;
    float sy = (i + 0.5f) * 0.5f - 0.5f;
    float sx = (j + 0.5f) * 0.5f - 0.5f;
    int y0 = (int)floorf(sy), x0 = (int)floorf(sx);
    float wy = sy - (float)y0, wx = sx - (float)x0;
    int y0c = min(max(y0, 0), 7),  y1c = min(max(y0 + 1, 0), 7);
    int x0c = min(max(x0, 0), 31), x1c = min(max(x0 + 1, 0), 31);
    float v00 = src[y0c * 32 + x0c], v01 = src[y0c * 32 + x1c];
    float v10 = src[y1c * 32 + x0c], v11 = src[y1c * 32 + x1c];
    buf[t] = v00 * (1.f - wy) * (1.f - wx) + v01 * (1.f - wy) * wx +
             v10 * wy * (1.f - wx) + v11 * wy * wx;
  }
  __syncthreads();
  float mn = 3.0e38f, mx = -3.0e38f;
  for (int t = threadIdx.x; t < 1024; t += 256) {
    mn = fminf(mn, buf[t]);
    mx = fmaxf(mx, buf[t]);
  }
  rmn[threadIdx.x] = mn;
  rmx[threadIdx.x] = mx;
  __syncthreads();
  for (int s = 128; s > 0; s >>= 1) {
    if ((int)threadIdx.x < s) {
      rmn[threadIdx.x] = fminf(rmn[threadIdx.x], rmn[threadIdx.x + s]);
      rmx[threadIdx.x] = fmaxf(rmx[threadIdx.x], rmx[threadIdx.x + s]);
    }
    __syncthreads();
  }
  float gmn = rmn[0];
  float inv = 1.f / (rmx[0] - gmn + 1e-8f);
  for (int t = threadIdx.x; t < 1024; t += 256)
    yout[bidx * 1024 + t] = (buf[t] - gmn) * inv;
}

// ---------------------------------------------------------------------------
// Host launcher
// ---------------------------------------------------------------------------
extern "C" void kernel_launch(void* const* d_in, const int* in_sizes, int n_in,
                              void* d_out, int out_size, void* d_ws, size_t ws_size,
                              hipStream_t stream) {
  (void)in_sizes; (void)n_in; (void)out_size; (void)ws_size;
  const float* x     = (const float*)d_in[0];
  const float* W_emb = (const float*)d_in[1];
  const float* b_emb = (const float*)d_in[2];
  const float* Wq = (const float*)d_in[3];  const float* bq = (const float*)d_in[4];
  const float* Wk = (const float*)d_in[5];  const float* bk = (const float*)d_in[6];
  const float* Wv = (const float*)d_in[7];  const float* bv = (const float*)d_in[8];
  const float* Wo = (const float*)d_in[9];  const float* bo = (const float*)d_in[10];
  const float* W_un = (const float*)d_in[11]; const float* b_un = (const float*)d_in[12];
  const float* W1 = (const float*)d_in[13]; const float* b1 = (const float*)d_in[14];
  const float* W2 = (const float*)d_in[15]; const float* b2 = (const float*)d_in[16];
  const float* W3 = (const float*)d_in[17]; const float* b3 = (const float*)d_in[18];

  // Workspace layout (regions reused as the pipeline advances):
  //   [0,16M)   : Hh fp16        -> later o_mid fp16 (same size)
  //   [16M,18M) : 4 transposed fp16 weights
  //   [18M,34M) : q fp16         \__ later att_out fp32 spans [18M,50M)
  //   [34M,50M) : k fp16         /
  //   [50M,66M) : vt fp16
  //   [66M,..)  : ff intermediates (fp32)
  char* ws = (char*)d_ws;
  _Float16* Hh    = (_Float16*)(ws);
  _Float16* Wqt   = (_Float16*)(ws + ((size_t)16 << 20));
  _Float16* Wkt   = Wqt + (size_t)DD * DD;
  _Float16* Wvt   = Wkt + (size_t)DD * DD;
  _Float16* Wot   = Wvt + (size_t)DD * DD;
  _Float16* qbuf  = (_Float16*)(ws + ((size_t)18 << 20));
  _Float16* kbuf  = (_Float16*)(ws + ((size_t)34 << 20));
  _Float16* vtbuf = (_Float16*)(ws + ((size_t)50 << 20));
  float*    att   = (float*)(ws + ((size_t)18 << 20));   // reuses q+k regions
  float*    y1    = (float*)(ws + ((size_t)66 << 20));
  float*    y2    = y1 + 32 * 2048;
  float*    y3    = y2 + 32 * 512;
  _Float16* o_mid = Hh;                                  // reuses Hh region

  float* yout    = (float*)d_out;           // y: [32,1,16,64] = 32768 floats
  float* x_atten = (float*)d_out + 32768;   // x_atten: [32,512,1] = 16384 floats

  // 1) weight transpose+convert (fp32 [in][out] -> fp16 [out][in])
  transpose_w_kernel<<<1024, 256, 0, stream>>>(Wq, Wqt);
  transpose_w_kernel<<<1024, 256, 0, stream>>>(Wk, Wkt);
  transpose_w_kernel<<<1024, 256, 0, stream>>>(Wv, Wvt);
  transpose_w_kernel<<<1024, 256, 0, stream>>>(Wo, Wot);

  // 2) embedding -> fp16 activations
  embed_kernel<<<32768, 256, 0, stream>>>(x, W_emb, b_emb, Hh);

  // 3) QKV projections (WMMA). 1/sqrt(DH)=0.125 folded into Q.
  wmma_gemm_qkvo<<<MROWS / 16, 256, 0, stream>>>(Hh, Wqt, bq, qbuf, 0, 0.125f);
  wmma_gemm_qkvo<<<MROWS / 16, 256, 0, stream>>>(Hh, Wkt, bk, kbuf, 1, 1.0f);
  wmma_gemm_qkvo<<<MROWS / 16, 256, 0, stream>>>(Hh, Wvt, bv, vtbuf, 2, 1.0f);

  // 4) attention (one wave per (b,h,16-query tile)); writes o_mid over Hh
  attention_kernel<<<BB * HH * (NN / 16), 32, 0, stream>>>(qbuf, kbuf, vtbuf, o_mid);

  // 5) output projection (WMMA) -> fp32
  wmma_gemm_qkvo<<<MROWS / 16, 256, 0, stream>>>(o_mid, Wot, bo, att, 3, 1.0f);

  // 6) unembed -> x_atten directly into d_out
  unembed_kernel<<<MROWS / 256, 256, 0, stream>>>(att, W_un, b_un, x_atten);

  // 7) ff chain (tiny)
  ff_gemm_kernel<<<(32 * 2048 + 255) / 256, 256, 0, stream>>>(x_atten, W1, b1, y1, 32, 512, 2048);
  ff_gemm_kernel<<<(32 * 512 + 255) / 256, 256, 0, stream>>>(y1, W2, b2, y2, 32, 2048, 512);
  ff_gemm_kernel<<<(32 * 256 + 255) / 256, 256, 0, stream>>>(y2, W3, b3, y3, 32, 512, 256);

  // 8) bilinear upsample + normalize -> y into d_out
  resize_norm_kernel<<<BB, 256, 0, stream>>>(y3, yout);
}
